// Scaled_Dot_Product_Attention_neg_77635828842739
// MI455X (gfx1250) — matrix operand order repaired and checked by reading, not compiled, not run on hardware
//
#include <hip/hip_runtime.h>

#define BB 32
#define LL 1024
#define DD 128

typedef __attribute__((ext_vector_type(16))) _Float16 v16h;
typedef __attribute__((ext_vector_type(8)))  float    v8f;

// padded LDS strides (avoid bank conflicts on row-gather)
#define QSTR 132   // f16 elems per Q row
#define SSTR 1040  // f32 elems per score row
#define BSTR 1034  // f16 elems per beta row
#define VSTR 66    // f16 elems per V^T row (64 j-values + pad), uint stride 33

__device__ __forceinline__ unsigned pk2(float a, float b) {
  auto h = __builtin_amdgcn_cvt_pkrtz(a, b);     // v_cvt_pk_rtz_f16_f32
  union { decltype(h) v; unsigned u; } cv; cv.v = h;
  return cv.u;
}

__global__ __launch_bounds__(256)
void attn2softmax_kernel(const float* __restrict__ Qg,
                         const float* __restrict__ Kg,
                         const float* __restrict__ Vg,
                         const float* __restrict__ scalep,
                         const float* __restrict__ kg,
                         float* __restrict__ Og)
{
  __shared__ _Float16 sQ[16 * QSTR];    //  4224 B
  __shared__ float    sS[16 * SSTR];    // 66560 B : full score rows
  __shared__ _Float16 sB[16 * BSTR];    // 33088 B : beta (unnormalized) f16
  __shared__ _Float16 sVt[DD * VSTR];   // 16896 B : V 64-row chunk, transposed
  __shared__ float    sRed[16 * 16];
  __shared__ float    sM[16];
  __shared__ float    sSum[16];
  __shared__ float    sInv2[16];

  const int tid   = threadIdx.x;
  const int lane  = tid & 31;
  const int wave  = tid >> 5;
  const int qt    = blockIdx.x;    // 0..63  (query tile)
  const int b     = blockIdx.y;    // 0..31  (batch)
  const int qbase = qt * 16;
  const float scale = *scalep;

  // ---- stage Q tile (16x128) into LDS as f16 ----
  {
    int row = tid >> 4;
    int col = (tid & 15) * 8;
    const float* qp = Qg + ((size_t)(b * LL + qbase + row)) * DD + col;
    float4 a = *(const float4*)(qp);
    float4 c = *(const float4*)(qp + 4);
    unsigned* qr = (unsigned*)(sQ + row * QSTR);
    qr[(col >> 1) + 0] = pk2(a.x, a.y);
    qr[(col >> 1) + 1] = pk2(a.z, a.w);
    qr[(col >> 1) + 2] = pk2(c.x, c.y);
    qr[(col >> 1) + 3] = pk2(c.z, c.w);
  }

  // prefetch the kg_sim rows this block streams later (64 KB tile)
  {
    const char* kgp = (const char*)(kg + ((size_t)(b * LL + qbase)) * LL);
    __builtin_prefetch(kgp + (size_t)tid * 128, 0, 1);
    __builtin_prefetch(kgp + 32768 + (size_t)tid * 128, 0, 1);
  }
  __syncthreads();

  const int m   = lane & 15;              // output row within tile (A/C layout)
  const int off = (lane >> 4) << 3;       // A-layout K offset: 0 or 8

  // ---- GEMM1: S = -scale * Q * K^T ; each wave owns 8 key tiles ----
  {
    // A operands depend only on k-step: hoist all 4 (32 VGPRs total)
    union { v16h v; unsigned u[8]; } Aq[4];
    const unsigned* qrow = (const unsigned*)(sQ + m * QSTR);
    #pragma unroll
    for (int ks = 0; ks < 4; ++ks) {
      const int cb = ks * 32;
      #pragma unroll
      for (int r = 0; r < 4; ++r) {
        Aq[ks].u[r]     = qrow[((cb + off) >> 1) + r];
        Aq[ks].u[4 + r] = qrow[((cb + 16 + off) >> 1) + r];
      }
    }
    #pragma clang loop unroll_count(2)
    for (int i = 0; i < 8; ++i) {
      const int kt   = wave * 8 + i;
      const int krow = b * LL + kt * 16 + (lane & 15);  // B-layout: N = lane%16
      v8f acc = {};
      #pragma unroll
      for (int ks = 0; ks < 4; ++ks) {
        const int cb = ks * 32;
        union { v16h v; unsigned u[8]; } Bm;
        // B element (K=d, N=keyrow): lanes>=16 take d offset +8 (table order)
        const float* kp = Kg + (size_t)krow * DD + cb + ((lane >> 4) << 3);
        float4 f0 = *(const float4*)(kp);
        float4 f1 = *(const float4*)(kp + 4);
        float4 f2 = *(const float4*)(kp + 16);
        float4 f3 = *(const float4*)(kp + 20);
        Bm.u[0] = pk2(f0.x, f0.y); Bm.u[1] = pk2(f0.z, f0.w);
        Bm.u[2] = pk2(f1.x, f1.y); Bm.u[3] = pk2(f1.z, f1.w);
        Bm.u[4] = pk2(f2.x, f2.y); Bm.u[5] = pk2(f2.z, f2.w);
        Bm.u[6] = pk2(f3.x, f3.y); Bm.u[7] = pk2(f3.z, f3.w);
        acc = __builtin_amdgcn_wmma_f32_16x16x32_f16(false, Aq[ks].v, false, Bm.v,
                                                     (short)0, acc, false, false);
      }
      #pragma unroll
      for (int r = 0; r < 8; ++r) {
        int mm = r + ((lane >> 4) << 3);   // C layout: rows r / r+8
        sS[mm * SSTR + kt * 16 + (lane & 15)] = -scale * acc[r];
      }
    }
  }
  __syncthreads();

  // ---- two softmaxes + kg_sim multiply, entirely out of LDS ----
  const int srow = tid >> 4;   // 16 threads per row
  const int jj0  = tid & 15;
  float* Srow = sS + srow * SSTR;

  { // row max of S
    float mx = -3.4e38f;
    for (int c = jj0; c < LL; c += 16) mx = fmaxf(mx, Srow[c]);
    sRed[srow * 16 + jj0] = mx;
  }
  __syncthreads();
  if (tid < 16) {
    float mx = sRed[tid * 16];
    for (int k = 1; k < 16; ++k) mx = fmaxf(mx, sRed[tid * 16 + k]);
    sM[tid] = mx;
  }
  __syncthreads();
  { // row sum of exp
    float m1 = sM[srow], sm = 0.f;
    for (int c = jj0; c < LL; c += 16) sm += __expf(Srow[c] - m1);
    sRed[srow * 16 + jj0] = sm;
  }
  __syncthreads();
  if (tid < 16) {
    float sm = 0.f;
    for (int k = 0; k < 16; ++k) sm += sRed[tid * 16 + k];
    sSum[tid] = sm;
  }
  __syncthreads();
  { // t = (-softmax(S)) * kg_sim ; stream kg rows coalesced; track new max
    float m1 = sM[srow];
    float inv1 = 1.0f / sSum[srow];
    const float* kgrow = kg + ((size_t)(b * LL + qbase + srow)) * LL;
    float mx2 = -3.4e38f;
    for (int c = jj0; c < LL; c += 16) {
      float a = -__expf(Srow[c] - m1) * inv1;
      float t = a * kgrow[c];
      Srow[c] = t;
      mx2 = fmaxf(mx2, t);
    }
    sRed[srow * 16 + jj0] = mx2;
  }
  __syncthreads();
  if (tid < 16) {
    float mx = sRed[tid * 16];
    for (int k = 1; k < 16; ++k) mx = fmaxf(mx, sRed[tid * 16 + k]);
    sM[tid] = mx;
  }
  __syncthreads();
  { // beta (unnormalized) -> f16 LDS; 1/sum folded into output store
    float m2 = sM[srow], sm = 0.f;
    _Float16* brow = sB + srow * BSTR;
    for (int c = jj0; c < LL; c += 16) {
      float e = __expf(Srow[c] - m2);
      brow[c] = (_Float16)e;
      sm += e;
    }
    sRed[srow * 16 + jj0] = sm;
  }
  __syncthreads();
  if (tid < 16) {
    float sm = 0.f;
    for (int k = 0; k < 16; ++k) sm += sRed[tid * 16 + k];
    sInv2[tid] = 1.0f / sm;
  }
  // visibility of sB/sInv2 guaranteed by barriers inside GEMM2 loop

  // ---- GEMM2: out = beta @ V ; each wave owns one 16-wide D tile ----
  v8f acc = {};
  const int dbase = wave * 16;
  const int joff  = (lane >> 4) << 4;   // unused; B load below uses off (K offset 0 or 8)
  unsigned* sVtU = (unsigned*)sVt;      // uint p at row d = {V[2p][d], V[2p+1][d]}
  const unsigned* brow = (const unsigned*)(sB + m * BSTR);

  #pragma clang loop unroll(disable)
  for (int it = 0; it < 16; ++it) {
    __syncthreads();
    { // stage V rows [64it, 64it+64): pack row pairs along j into uints
      int pr   = tid >> 3;            // j-pair 0..31
      int col0 = (tid & 7) * 16;      // d columns
      const float* vp = Vg + ((size_t)(b * LL + it * 64 + 2 * pr)) * DD + col0;
      float4 a0 = *(const float4*)(vp);
      float4 a1 = *(const float4*)(vp + 4);
      float4 a2 = *(const float4*)(vp + 8);
      float4 a3 = *(const float4*)(vp + 12);
      float4 b0 = *(const float4*)(vp + DD);
      float4 b1 = *(const float4*)(vp + DD + 4);
      float4 b2 = *(const float4*)(vp + DD + 8);
      float4 b3 = *(const float4*)(vp + DD + 12);
      float va[16] = {a0.x,a0.y,a0.z,a0.w, a1.x,a1.y,a1.z,a1.w,
                      a2.x,a2.y,a2.z,a2.w, a3.x,a3.y,a3.z,a3.w};
      float vb[16] = {b0.x,b0.y,b0.z,b0.w, b1.x,b1.y,b1.z,b1.w,
                      b2.x,b2.y,b2.z,b2.w, b3.x,b3.y,b3.z,b3.w};
      #pragma unroll
      for (int c = 0; c < 16; ++c)
        sVtU[(col0 + c) * (VSTR / 2) + pr] = pk2(va[c], vb[c]);
    }
    __syncthreads();
    const unsigned* vrow = sVtU + (dbase + (lane & 15)) * (VSTR / 2);
    #pragma unroll
    for (int sub = 0; sub < 64; sub += 32) {
      union { v16h v; unsigned u[8]; } A, Bm;
      const int cb = it * 64 + sub;
      #pragma unroll
      for (int r = 0; r < 4; ++r) {
        A.u[r]     = brow[((cb + off) >> 1) + r];
        A.u[4 + r] = brow[((cb + 16 + off) >> 1) + r];
      }
      #pragma unroll
      for (int r = 0; r < 4; ++r) { Bm.u[r] = vrow[((sub + off) >> 1) + r]; Bm.u[4 + r] = vrow[((sub + 16 + off) >> 1) + r]; }
      acc = __builtin_amdgcn_wmma_f32_16x16x32_f16(false, A.v, false, Bm.v,
                                                   (short)0, acc, false, false);
    }
  }

  #pragma unroll
  for (int r = 0; r < 8; ++r) {
    int mm = r + ((lane >> 4) << 3);
    Og[((size_t)(b * LL + qbase + mm)) * DD + dbase + (lane & 15)] =
        acc[r] * sInv2[mm];
  }
}

extern "C" void kernel_launch(void* const* d_in, const int* in_sizes, int n_in,
                              void* d_out, int out_size, void* d_ws, size_t ws_size,
                              hipStream_t stream) {
  const float* Q     = (const float*)d_in[0];
  const float* K     = (const float*)d_in[1];
  const float* V     = (const float*)d_in[2];
  const float* scale = (const float*)d_in[3];
  const float* kg    = (const float*)d_in[4];
  float* out = (float*)d_out;
  dim3 grid(LL / 16, BB);
  attn2softmax_kernel<<<grid, dim3(256), 0, stream>>>(Q, K, V, scale, kg, out);
}
